// CPELite_27006754357915
// MI455X (gfx1250) — matrix operand order repaired and checked by reading, not compiled, not run on hardware
//
#include <hip/hip_runtime.h>

typedef __attribute__((ext_vector_type(16))) _Float16 v16h;
typedef __attribute__((ext_vector_type(8)))  _Float16 v8h;
typedef __attribute__((ext_vector_type(8)))  float    v8f;
typedef __attribute__((ext_vector_type(4)))  float    v4f;

#define NPATCH 4096
#define WPB    8          // waves per block (256 threads)
#define LN_EPS 1e-5f

// ---------------- kernel 0: zero segment-sum workspace ----------------
__global__ void cpe_zero(float* ws, int n) {
    int i = blockIdx.x * blockDim.x + threadIdx.x;
    if (i < n) ws[i] = 0.0f;
}

// ---------------- kernel 1: segment sums (coords + count) -------------
__global__ void cpe_scatter(const float* __restrict__ coord,
                            const int* __restrict__ pid,
                            float* __restrict__ sums, int n) {
    int i = blockIdx.x * blockDim.x + threadIdx.x;
    if (i < n) {
        int p = pid[i];
        atomicAdd(&sums[p * 4 + 0], coord[i * 3 + 0]);
        atomicAdd(&sums[p * 4 + 1], coord[i * 3 + 1]);
        atomicAdd(&sums[p * 4 + 2], coord[i * 3 + 2]);
        atomicAdd(&sums[p * 4 + 3], 1.0f);
    }
}

// ---------------- kernel 2: centers = sums / max(cnt,1) ---------------
__global__ void cpe_centers(const float4* __restrict__ sums,
                            float4* __restrict__ centers) {
    int p = blockIdx.x * blockDim.x + threadIdx.x;
    if (p < NPATCH) {
        float4 s = sums[p];
        float inv = 1.0f / fmaxf(s.w, 1.0f);
        centers[p] = make_float4(s.x * inv, s.y * inv, s.z * inv, 0.0f);
    }
}

__device__ __forceinline__ v8f load8(const float* p) {
    v4f a = *(const v4f*)p;
    v4f b = *(const v4f*)(p + 4);
    v8f r;
    r[0]=a[0]; r[1]=a[1]; r[2]=a[2]; r[3]=a[3];
    r[4]=b[0]; r[5]=b[1]; r[6]=b[2]; r[7]=b[3];
    return r;
}

// ---------------- kernel 3: fused MLP + residual + PW-GEMM + LN -------
// One wave32 owns a 16-row tile. B-fragments for both 64x64 GEMMs live in
// LDS in per-lane WMMA layout. Both GEMMs accumulate from C=0 (inline
// constant); biases are applied in the cheap elementwise stages. Residual
// x is loaded in two 16-wide batches after GEMM1 so nothing bulky stays
// live across the matrix ops (avoids scratch spills).
// NOTE: assumes n % 16 == 0 (true for the N=1M harness).
__global__ __launch_bounds__(256, 4)
void cpe_main(const float* __restrict__ x, const float* __restrict__ coord,
              const int* __restrict__ pid, const float* __restrict__ W1,
              const float* __restrict__ b1, const float* __restrict__ W2,
              const float* __restrict__ b2, const float* __restrict__ dw_w,
              const float* __restrict__ dw_b, const float* __restrict__ pwW,
              const float* __restrict__ pw_b, const float* __restrict__ ln_g,
              const float* __restrict__ ln_b,
              const float4* __restrict__ centers,
              float* __restrict__ out, int nTiles) {
    // [gemm][kt][nt][lane][e] halves: 2*2*4*32*16*2B = 16KB
    __shared__ alignas(32) _Float16 ldsW[2 * 2 * 4 * 32 * 16];
    __shared__ alignas(32) _Float16 ldsY[WPB][16 * 64];  // 2KB/wave transpose buf

    const int lane = threadIdx.x & 31;
    const int wave = threadIdx.x >> 5;
    const int c16  = lane & 15;   // column / A-row index
    const int hg   = lane >> 4;   // lane half-group
    _Float16* myY  = ldsY[wave];

    // ---- wave 0 packs B fragments (V_WMMA 32x16 f16 B layout) into LDS:
    //      lane half hg holds K = kt*32 + (e>>3)*16 + hg*8 + (e&7), column = nt*16 + c16
    if (wave == 0) {
        #pragma unroll
        for (int kt = 0; kt < 2; kt++)
            #pragma unroll
            for (int nt = 0; nt < 4; nt++) {
                int nn = nt * 16 + c16;
                _Float16* w2d = ldsW + (((0 * 2 + kt) * 4 + nt) * 32 + lane) * 16;
                _Float16* pwd = ldsW + (((1 * 2 + kt) * 4 + nt) * 32 + lane) * 16;
                #pragma unroll
                for (int e = 0; e < 16; e++) {
                    int k = kt * 32 + (e >> 3) * 16 + hg * 8 + (e & 7);
                    w2d[e] = (_Float16)W2[k * 64 + nn];
                    // fold depthwise per-channel scale into pointwise weights
                    pwd[e] = (_Float16)(dw_w[k] * pwW[k * 64 + nn]);
                }
            }
    }
    // per-column constants (bias2, folded pw bias, layernorm affine) in VGPRs
    float b2v[4], bPv[4], lgv[4], lbv[4];
    #pragma unroll
    for (int nt = 0; nt < 4; nt++) {
        int nn = nt * 16 + c16;
        b2v[nt] = b2[nn];
        float s = pw_b[nn];
        for (int k = 0; k < 64; k++) s = fmaf(dw_b[k], pwW[k * 64 + nn], s);
        bPv[nt] = s;
        lgv[nt] = ln_g[nn];
        lbv[nt] = ln_b[nn];
    }
    __syncthreads();

    const _Float16* wfrag = ldsW + lane * 16;     // + fragIdx*512 (immediate)
    const int waveGlobal  = blockIdx.x * WPB + wave;
    const int waveStride  = gridDim.x * WPB;

    for (int tile = waveGlobal; tile < nTiles; tile += waveStride) {
        const int rowBase = tile * 16;
        // per-lane base pointers: all element addressing via immediate offsets
        const float*    xb  = x   + (size_t)rowBase * 64 + hg * 512 + c16;
        float*          ob  = out + (size_t)rowBase * 64 + hg * 512 + c16;
        _Float16*       yst = myY + hg * 512 + c16;     // D-layout staging store
        const _Float16* yld = myY + c16 * 64 + hg * 8;  // A-layout staging load

        // prefetch next tile's x rows (speculative global_prefetch)
        int nextTile = tile + waveStride;
        if (nextTile < nTiles) {
            const char* pf = (const char*)(x + (size_t)nextTile * 16 * 64) + lane * 128;
            __builtin_prefetch(pf, 0, 0);
        }

        // ---- delta for this lane's A-row (row = rowBase + c16)
        int row = rowBase + c16;
        int p = pid[row];
        float4 ctr = centers[p];
        float dx = coord[row * 3 + 0] - ctr.x;
        float dy = coord[row * 3 + 1] - ctr.y;
        float dz = coord[row * 3 + 2] - ctr.z;

        // ---- GEMM1 A fragments built directly: h = gelu(delta@W1 + b1)
        //      A layout (16-bit, 16x32): lane row = c16,
        //      K(e<8) = kt*32 + hg*8 + e ; K(e>=8) = kt*32 + 16 + hg*8 + (e-8)
        v16h A1[2];
        #pragma unroll
        for (int kt = 0; kt < 2; kt++) {
            v16h a;
            #pragma unroll
            for (int hh = 0; hh < 2; hh++) {
                int k0 = kt * 32 + hh * 16 + hg * 8;
                v8f r0 = load8(W1 + k0);
                v8f r1 = load8(W1 + 64 + k0);
                v8f r2 = load8(W1 + 128 + k0);
                v8f bb = load8(b1 + k0);
                #pragma unroll
                for (int j = 0; j < 8; j++) {
                    float v = fmaf(dx, r0[j], fmaf(dy, r1[j], fmaf(dz, r2[j], bb[j])));
                    v = 0.5f * v * (1.0f + erff(v * 0.70710678118654752f)); // exact GELU
                    a[hh * 8 + j] = (_Float16)v;
                }
            }
            A1[kt] = a;
        }

        // ---- GEMM1: pos = h @ W2 (C starts at zero); B from LDS
        v8f acc[4];
        #pragma unroll
        for (int nt = 0; nt < 4; nt++) {
            v8f c = {};
            v16h B0 = *(const v16h*)(wfrag + (0 * 4 + nt) * 512);
            v16h B1 = *(const v16h*)(wfrag + (1 * 4 + nt) * 512);
            c = __builtin_amdgcn_wmma_f32_16x16x32_f16(false, A1[0], false, B0, (short)0, c, false, false);
            c = __builtin_amdgcn_wmma_f32_16x16x32_f16(false, A1[1], false, B1, (short)0, c, false, false);
            acc[nt] = c;
        }

        // ---- residual + b2 in D layout, stage t (f16) for the transpose.
        //      Two batches of 16 loads: enough in flight to hide latency,
        //      few enough to avoid spilling across the GEMMs.
        #pragma unroll
        for (int p2 = 0; p2 < 2; p2++) {
            float xv[16];
            #pragma unroll
            for (int nt2 = 0; nt2 < 2; nt2++)
                #pragma unroll
                for (int r = 0; r < 8; r++)
                    xv[nt2 * 8 + r] = xb[r * 64 + (p2 * 2 + nt2) * 16];
            #pragma unroll
            for (int nt2 = 0; nt2 < 2; nt2++) {
                int nt = p2 * 2 + nt2;
                #pragma unroll
                for (int r = 0; r < 8; r++)
                    yst[r * 64 + nt * 16] =
                        (_Float16)(xv[nt2 * 8 + r] + acc[nt][r] + b2v[nt]);
            }
        }
        asm volatile("s_wait_dscnt 0" ::: "memory");

        // ---- reload t as GEMM2 A fragments (ds_load_b128 chunks)
        v16h A2[2];
        #pragma unroll
        for (int kt = 0; kt < 2; kt++) {
            v16h a;
            #pragma unroll
            for (int hh = 0; hh < 2; hh++) {
                v8h chunk = *(const v8h*)(yld + kt * 32 + hh * 16);
                #pragma unroll
                for (int j = 0; j < 8; j++) a[hh * 8 + j] = chunk[j];
            }
            A2[kt] = a;
        }

        // ---- GEMM2: o = t @ (dw_w*pw_W) (C starts at zero); B from LDS
        #pragma unroll
        for (int nt = 0; nt < 4; nt++) {
            v8f c = {};
            v16h B0 = *(const v16h*)(wfrag + (4 + 0 * 4 + nt) * 512);
            v16h B1 = *(const v16h*)(wfrag + (4 + 1 * 4 + nt) * 512);
            c = __builtin_amdgcn_wmma_f32_16x16x32_f16(false, A2[0], false, B0, (short)0, c, false, false);
            c = __builtin_amdgcn_wmma_f32_16x16x32_f16(false, A2[1], false, B1, (short)0, c, false, false);
            acc[nt] = c;
        }

        // ---- LayerNorm per row (folded pw bias added pre-stats), then store
        #pragma unroll
        for (int r = 0; r < 8; r++) {
            float v0 = acc[0][r] + bPv[0];
            float v1 = acc[1][r] + bPv[1];
            float v2 = acc[2][r] + bPv[2];
            float v3 = acc[3][r] + bPv[3];
            float s1 = v0 + v1 + v2 + v3;
            float s2 = v0 * v0 + v1 * v1 + v2 * v2 + v3 * v3;
            #pragma unroll
            for (int mask = 8; mask >= 1; mask >>= 1) {
                s1 += __shfl_xor(s1, mask, 32);
                s2 += __shfl_xor(s2, mask, 32);
            }
            float mu  = s1 * (1.0f / 64.0f);
            float var = s2 * (1.0f / 64.0f) - mu * mu;
            float inv = rsqrtf(var + LN_EPS);
            ob[r * 64 + 0]  = (v0 - mu) * inv * lgv[0] + lbv[0];
            ob[r * 64 + 16] = (v1 - mu) * inv * lgv[1] + lbv[1];
            ob[r * 64 + 32] = (v2 - mu) * inv * lgv[2] + lbv[2];
            ob[r * 64 + 48] = (v3 - mu) * inv * lgv[3] + lbv[3];
        }
    }
}

extern "C" void kernel_launch(void* const* d_in, const int* in_sizes, int n_in,
                              void* d_out, int out_size, void* d_ws, size_t ws_size,
                              hipStream_t stream) {
    const float* x     = (const float*)d_in[0];
    const float* coord = (const float*)d_in[1];
    const int*   pids  = (const int*)d_in[2];
    const float* W1    = (const float*)d_in[3];
    const float* b1    = (const float*)d_in[4];
    const float* W2    = (const float*)d_in[5];
    const float* b2    = (const float*)d_in[6];
    const float* dw_w  = (const float*)d_in[7];
    const float* dw_b  = (const float*)d_in[8];
    const float* pwW   = (const float*)d_in[9];
    const float* pw_b  = (const float*)d_in[10];
    const float* ln_g  = (const float*)d_in[11];
    const float* ln_b  = (const float*)d_in[12];
    float* out = (float*)d_out;

    const int n = in_sizes[0] / 64;  // N rows (1,000,000: divisible by 16)

    float*  sums    = (float*)d_ws;                              // P * float4
    float4* centers = (float4*)((char*)d_ws + NPATCH * 4 * sizeof(float));

    // 0) zero segment sums (every launch: graph replays re-accumulate)
    {
        int tot = NPATCH * 4;
        cpe_zero<<<(tot + 255) / 256, 256, 0, stream>>>(sums, tot);
    }
    // 1) segment sums via fp32 atomics
    cpe_scatter<<<(n + 255) / 256, 256, 0, stream>>>(coord, pids, sums, n);
    // 2) centers
    cpe_centers<<<(NPATCH + 255) / 256, 256, 0, stream>>>((const float4*)sums, centers);
    // 3) fused main pipeline: 16-row tiles, WMMA GEMMs
    {
        int nTiles = n / 16;
        int blocks = 512;  // 4096 waves, grid-stride over 62500 tiles
        cpe_main<<<blocks, 256, 0, stream>>>(x, coord, pids, W1, b1, W2, b2,
                                             dw_w, dw_b, pwW, pw_b, ln_g, ln_b,
                                             centers, out, nTiles);
    }
}
